// TriangleMultiplicationOutgoing_317827580832
// MI455X (gfx1250) — compile-verified, matrix operand order checked by reading, not run on hardware
//
#include <hip/hip_runtime.h>
#include <math.h>

// ---------------------------------------------------------------------------
// TriangleMultiplicationOutgoing for MI455X (gfx1250, wave32, WMMA)
//   N = 512 sequence, D = 128 channels
//   Dominant cost: t[i,j,d] = sum_k a[i,k,d]*b[j,k,d]  == 128 x (512^3 GEMM)
//   -> bf16 v_wmma_f32_16x16x32_bf16, f32 accumulate.
// ---------------------------------------------------------------------------

typedef __attribute__((ext_vector_type(16))) __bf16 v16bf;
typedef __attribute__((ext_vector_type(8)))  float  v8f;

#define NN   512
#define DD   128
#define NN2  (NN * NN)          // 262144 positions
#define EPSV 1e-5f

union V16U { v16bf v; uint4 q[2]; };

// Load one 16x32 bf16 WMMA operand tile (A or B; both use lane = 16-dim index,
// 32 K values packed per lane-half) from row-major memory with given row stride.
// ISA 16-bit operand layout: lanes 0-15 hold K=[0..7] then K=[16..23];
// lanes 16-31 hold K=[8..15] then K=[24..31].
__device__ __forceinline__ v16bf ld_op(const __bf16* base, int row0, int k0,
                                       int rowstride, int lane) {
  const __bf16* p = base + (size_t)(row0 + (lane & 15)) * rowstride
                         + (k0 + ((lane >> 4) << 3));
  V16U u;
  u.q[0] = *(const uint4*)p;          // K .. K+7
  u.q[1] = *(const uint4*)(p + 16);   // K+16 .. K+23
  return u.v;
}

__device__ __forceinline__ v8f vzero8() {
  v8f z = {0.f, 0.f, 0.f, 0.f, 0.f, 0.f, 0.f, 0.f};
  return z;
}

__device__ __forceinline__ float wave_sum32(float v) {
  for (int off = 16; off > 0; off >>= 1) v += __shfl_xor(v, off, 32);
  return v;
}

__device__ __forceinline__ float sigmoidf(float x) {
  return 1.0f / (1.0f + __expf(-x));
}

__device__ __forceinline__ v8f wmma_bf16(v16bf a, v16bf b, v8f c) {
  return __builtin_amdgcn_wmma_f32_16x16x32_bf16(
      /*neg_a=*/false, a, /*neg_b=*/false, b,
      /*c_mod=*/(short)0, c, /*reuse_a=*/false, /*reuse_b=*/false);
}

// ---------------------------------------------------------------------------
// Kernel 0: convert weights to bf16.
//   wcatB[640][128] = concat(p_in_w[256][128], g_in_w[256][128], g_out_w[128][128])
//   wpoB [128][128] = p_out_w
// ---------------------------------------------------------------------------
__global__ void __launch_bounds__(256) k_prep(const float* __restrict__ p_in_w,
                                              const float* __restrict__ g_in_w,
                                              const float* __restrict__ g_out_w,
                                              const float* __restrict__ p_out_w,
                                              __bf16* __restrict__ wcatB,
                                              __bf16* __restrict__ wpoB) {
  int i = blockIdx.x * 256 + threadIdx.x;
  const int WCAT = 640 * 128;          // 81920
  if (i < WCAT) {
    int e = i >> 7;
    float v;
    if (e < 256)      v = p_in_w[i];
    else if (e < 512) v = g_in_w[i - 256 * 128];
    else              v = g_out_w[i - 512 * 128];
    wcatB[i] = (__bf16)v;
  } else if (i < WCAT + 128 * 128) {
    wpoB[i - WCAT] = (__bf16)p_out_w[i - WCAT];
  }
}

// ---------------------------------------------------------------------------
// Kernel 1: LN(x) -> WMMA projections p,g,gate -> h = p*sigmoid(g)*mask
//   writes aT[d][pos] = h[pos][d]      (d <  128)  bf16
//          bT[d][pos] = h[pos][d+128]  (d >= 128)  bf16
//          gate[pos][e] = sigmoid(xn @ g_out_w^T)  f32
// Block: 256 threads = 8 waves; each wave owns 64 consecutive positions
// (4 WMMA M-tiles of 16), GEMM K = 128 (4 k-steps), N tiles from wcatB.
// ---------------------------------------------------------------------------
__global__ void __launch_bounds__(256) k_pass1(const float* __restrict__ x,
                                               const float* __restrict__ mask,
                                               const float* __restrict__ niw,
                                               const float* __restrict__ nib,
                                               const __bf16* __restrict__ wcatB,
                                               __bf16* __restrict__ aT,
                                               __bf16* __restrict__ bT,
                                               float* __restrict__ gate) {
  __shared__ __align__(16) __bf16 xnbuf[8][64][128];   // 128 KB
  const int lane = threadIdx.x & 31;
  const int wave = threadIdx.x >> 5;
  const int posbase = (blockIdx.x * 8 + wave) * 64;

  float4 w4 = *(const float4*)(niw + lane * 4);
  float4 b4 = *(const float4*)(nib + lane * 4);

  // --- LayerNorm 64 positions, bf16 result into LDS (wave-private buffer) ---
  for (int p = 0; p < 64; ++p) {
    const int pos = posbase + p;
    float4 xv = *(const float4*)(x + (size_t)pos * DD + lane * 4);
    float mu = wave_sum32(xv.x + xv.y + xv.z + xv.w) * (1.0f / DD);
    float d0 = xv.x - mu, d1 = xv.y - mu, d2 = xv.z - mu, d3 = xv.w - mu;
    float var = wave_sum32(d0 * d0 + d1 * d1 + d2 * d2 + d3 * d3) * (1.0f / DD);
    float rstd = rsqrtf(var + EPSV);
    __bf16* row = &xnbuf[wave][p][lane * 4];
    row[0] = (__bf16)(d0 * rstd * w4.x + b4.x);
    row[1] = (__bf16)(d1 * rstd * w4.y + b4.y);
    row[2] = (__bf16)(d2 * rstd * w4.z + b4.z);
    row[3] = (__bf16)(d3 * rstd * w4.w + b4.w);
  }

  const __bf16* xnb = &xnbuf[wave][0][0];
  const int ecol = lane & 15;
  const int mrow = (lane >> 4) << 3;

  // --- paired p/g tiles: t in [0,16); p channels e=t*16.., g channels +256 ---
  for (int t = 0; t < 16; ++t) {
    v8f accp[4], accg[4];
#pragma unroll
    for (int mi = 0; mi < 4; ++mi) { accp[mi] = vzero8(); accg[mi] = vzero8(); }
#pragma unroll
    for (int ks = 0; ks < 4; ++ks) {
      v16bf bp = ld_op(wcatB, t * 16, ks * 32, DD, lane);
      v16bf bg = ld_op(wcatB, 256 + t * 16, ks * 32, DD, lane);
#pragma unroll
      for (int mi = 0; mi < 4; ++mi) {
        v16bf am = ld_op(xnb, mi * 16, ks * 32, DD, lane);
        accp[mi] = wmma_bf16(am, bp, accp[mi]);
        accg[mi] = wmma_bf16(am, bg, accg[mi]);
      }
    }
    const int dch = t * 16 + ecol;                        // 0..255
    __bf16* dst = (t < 8) ? (aT + (size_t)dch * NN2)
                          : (bT + (size_t)(dch - 128) * NN2);
#pragma unroll
    for (int mi = 0; mi < 4; ++mi) {
#pragma unroll
      for (int v = 0; v < 8; ++v) {
        const int pos = posbase + mi * 16 + mrow + v;
        float h = accp[mi][v] * sigmoidf(accg[mi][v]) * mask[pos];
        dst[pos] = (__bf16)h;
      }
    }
  }

  // --- gate tiles: sigmoid(xn @ g_out_w^T), channels at wcatB rows 512.. ---
  for (int g = 0; g < 8; ++g) {
    v8f acc[4];
#pragma unroll
    for (int mi = 0; mi < 4; ++mi) acc[mi] = vzero8();
#pragma unroll
    for (int ks = 0; ks < 4; ++ks) {
      v16bf bg = ld_op(wcatB, 512 + g * 16, ks * 32, DD, lane);
#pragma unroll
      for (int mi = 0; mi < 4; ++mi) {
        v16bf am = ld_op(xnb, mi * 16, ks * 32, DD, lane);
        acc[mi] = wmma_bf16(am, bg, acc[mi]);
      }
    }
    const int e = g * 16 + ecol;
#pragma unroll
    for (int mi = 0; mi < 4; ++mi) {
#pragma unroll
      for (int v = 0; v < 8; ++v) {
        const int pos = posbase + mi * 16 + mrow + v;
        gate[(size_t)pos * DD + e] = sigmoidf(acc[mi][v]);
      }
    }
  }
}

// ---------------------------------------------------------------------------
// Kernel 2: the triangle GEMMs.  For each d: tT[d] = A_d (512x512) * B_d^T.
// Block = 4 waves; each wave computes a 64x32 f32 tile for one d channel
// (8 WMMAs per K-step, K=512 in steps of 32).
// Grid: x = 16 j-tiles(32), y = 2 i-bands(256), z = 128 d.
// ---------------------------------------------------------------------------
__global__ void __launch_bounds__(128) k_pass2(const __bf16* __restrict__ aT,
                                               const __bf16* __restrict__ bT,
                                               float* __restrict__ tT) {
  const int lane = threadIdx.x & 31;
  const int wave = threadIdx.x >> 5;
  const int j0 = blockIdx.x * 32;
  const int i0 = blockIdx.y * 256 + wave * 64;
  const int d  = blockIdx.z;

  const __bf16* A = aT + (size_t)d * NN2;   // [i][k], k contiguous
  const __bf16* B = bT + (size_t)d * NN2;   // [j][k], k contiguous

  v8f acc[4][2];
#pragma unroll
  for (int mi = 0; mi < 4; ++mi)
#pragma unroll
    for (int nj = 0; nj < 2; ++nj) acc[mi][nj] = vzero8();

  for (int k0 = 0; k0 < NN; k0 += 32) {
    v16bf bm0 = ld_op(B, j0,      k0, NN, lane);
    v16bf bm1 = ld_op(B, j0 + 16, k0, NN, lane);
#pragma unroll
    for (int mi = 0; mi < 4; ++mi) {
      v16bf am = ld_op(A, i0 + 16 * mi, k0, NN, lane);
      acc[mi][0] = wmma_bf16(am, bm0, acc[mi][0]);
      acc[mi][1] = wmma_bf16(am, bm1, acc[mi][1]);
    }
  }

  float* out = tT + (size_t)d * NN2;        // tT layout: [d][i][j]
  const int nlo = lane & 15;
  const int mhi = (lane >> 4) << 3;
#pragma unroll
  for (int mi = 0; mi < 4; ++mi)
#pragma unroll
    for (int nj = 0; nj < 2; ++nj)
#pragma unroll
      for (int v = 0; v < 8; ++v)
        out[(size_t)(i0 + 16 * mi + mhi + v) * NN + j0 + 16 * nj + nlo] =
            acc[mi][nj][v];
}

// ---------------------------------------------------------------------------
// Kernel 3: out = LN(t) @ p_out_w^T * gate
// Same structure as pass1: 8 waves x 64 positions, K=128, N=128 (8 tiles).
// Gathers t from d-major slabs (full cacheline reuse across the 64-pos sweep).
// ---------------------------------------------------------------------------
__global__ void __launch_bounds__(256) k_pass3(const float* __restrict__ tT,
                                               const float* __restrict__ gate,
                                               const float* __restrict__ now,
                                               const float* __restrict__ nob,
                                               const __bf16* __restrict__ wpoB,
                                               float* __restrict__ out) {
  __shared__ __align__(16) __bf16 tnbuf[8][64][128];   // 128 KB
  const int lane = threadIdx.x & 31;
  const int wave = threadIdx.x >> 5;
  const int posbase = (blockIdx.x * 8 + wave) * 64;

  float4 w4 = *(const float4*)(now + lane * 4);
  float4 b4 = *(const float4*)(nob + lane * 4);

  for (int p = 0; p < 64; ++p) {
    const int pos = posbase + p;
    float tv0 = tT[(size_t)(lane * 4 + 0) * NN2 + pos];
    float tv1 = tT[(size_t)(lane * 4 + 1) * NN2 + pos];
    float tv2 = tT[(size_t)(lane * 4 + 2) * NN2 + pos];
    float tv3 = tT[(size_t)(lane * 4 + 3) * NN2 + pos];
    float mu = wave_sum32(tv0 + tv1 + tv2 + tv3) * (1.0f / DD);
    float d0 = tv0 - mu, d1 = tv1 - mu, d2 = tv2 - mu, d3 = tv3 - mu;
    float var = wave_sum32(d0 * d0 + d1 * d1 + d2 * d2 + d3 * d3) * (1.0f / DD);
    float rstd = rsqrtf(var + EPSV);
    __bf16* row = &tnbuf[wave][p][lane * 4];
    row[0] = (__bf16)(d0 * rstd * w4.x + b4.x);
    row[1] = (__bf16)(d1 * rstd * w4.y + b4.y);
    row[2] = (__bf16)(d2 * rstd * w4.z + b4.z);
    row[3] = (__bf16)(d3 * rstd * w4.w + b4.w);
  }

  const __bf16* tnb = &tnbuf[wave][0][0];
  const int ecol = lane & 15;
  const int mrow = (lane >> 4) << 3;

  for (int nt = 0; nt < 8; ++nt) {
    v8f acc[4];
#pragma unroll
    for (int mi = 0; mi < 4; ++mi) acc[mi] = vzero8();
#pragma unroll
    for (int ks = 0; ks < 4; ++ks) {
      v16bf bw = ld_op(wpoB, nt * 16, ks * 32, DD, lane);
#pragma unroll
      for (int mi = 0; mi < 4; ++mi) {
        v16bf am = ld_op(tnb, mi * 16, ks * 32, DD, lane);
        acc[mi] = wmma_bf16(am, bw, acc[mi]);
      }
    }
    const int e = nt * 16 + ecol;
#pragma unroll
    for (int mi = 0; mi < 4; ++mi) {
#pragma unroll
      for (int v = 0; v < 8; ++v) {
        const int pos = posbase + mi * 16 + mrow + v;
        out[(size_t)pos * DD + e] = acc[mi][v] * gate[(size_t)pos * DD + e];
      }
    }
  }
}

// ---------------------------------------------------------------------------
// Host launcher. Workspace layout (403 MB total):
//   wcatB(160KB) | wpoB(32KB) | aT(64MB) | bT(64MB) | tT(128MB) | gate(128MB)
// ---------------------------------------------------------------------------
extern "C" void kernel_launch(void* const* d_in, const int* in_sizes, int n_in,
                              void* d_out, int out_size, void* d_ws, size_t ws_size,
                              hipStream_t stream) {
  const float* x       = (const float*)d_in[0];
  const float* mask    = (const float*)d_in[1];
  const float* niw     = (const float*)d_in[2];
  const float* nib     = (const float*)d_in[3];
  const float* p_in_w  = (const float*)d_in[4];
  const float* g_in_w  = (const float*)d_in[5];
  const float* now     = (const float*)d_in[6];
  const float* nob     = (const float*)d_in[7];
  const float* p_out_w = (const float*)d_in[8];
  const float* g_out_w = (const float*)d_in[9];

  char* ws = (char*)d_ws;
  size_t off = 0;
  __bf16* wcatB = (__bf16*)(ws + off); off += (size_t)640 * 128 * 2;          // 163840
  __bf16* wpoB  = (__bf16*)(ws + off); off += (size_t)128 * 128 * 2;          // +32768
  __bf16* aT    = (__bf16*)(ws + off); off += (size_t)DD * NN2 * 2;           // +64MB
  __bf16* bT    = (__bf16*)(ws + off); off += (size_t)DD * NN2 * 2;           // +64MB
  float*  tT    = (float*) (ws + off); off += (size_t)DD * NN2 * 4;           // +128MB
  float*  gate  = (float*) (ws + off);                                        // +128MB

  k_prep<<<384, 256, 0, stream>>>(p_in_w, g_in_w, g_out_w, p_out_w, wcatB, wpoB);
  k_pass1<<<512, 256, 0, stream>>>(x, mask, niw, nib, wcatB, aT, bT, gate);
  k_pass2<<<dim3(16, 2, 128), 128, 0, stream>>>(aT, bT, tT);
  k_pass3<<<512, 256, 0, stream>>>(tT, gate, now, nob, wpoB, (float*)d_out);
}
